// PyTorchTransformerLayer_58471684768096
// MI455X (gfx1250) — matrix operand order repaired and checked by reading, not compiled, not run on hardware
//
#include <hip/hip_runtime.h>
#include <cstdint>

// ---- problem constants -----------------------------------------------------
#define B_    2
#define T_    2048
#define D_    2048
#define H_    16
#define HKV_  4
#define HD_   128
#define I_    5632
#define BT_   (B_ * T_)        // 4096 rows of activations
#define QD_   (H_ * HD_)       // 2048
#define KVD_  (HKV_ * HD_)     // 512

typedef __attribute__((ext_vector_type(16))) _Float16 v16h;
typedef __attribute__((ext_vector_type(8)))  float    v8f;
typedef unsigned int u32;
typedef unsigned long long u64;
typedef __attribute__((ext_vector_type(4))) u32 v4u;
typedef __attribute__((ext_vector_type(4))) int v4i;
typedef __attribute__((ext_vector_type(8))) int v8i;

union AF  { v16h v; u32 u[8]; _Float16 h[16]; };
union BFr { v16h v; u32 u[8]; _Float16 h[16]; };
union CF  { v8f  v; float f[8]; };

__device__ __forceinline__ v8f wmma16(v16h a, v16h b, v8f c) {
  // D = A(16x32 f16) * B(32x16 f16) + C(16x16 f32)
  return __builtin_amdgcn_wmma_f32_16x16x32_f16(false, a, false, b,
                                                (short)0, c, false, false);
}

// ---- Tensor Data Mover: async 2D tile load Global -> LDS -------------------
#if defined(__has_builtin)
#if __has_builtin(__builtin_amdgcn_tensor_load_to_lds) && \
    __has_builtin(__builtin_amdgcn_s_wait_tensorcnt)
#define HAVE_TDM 1
#endif
#endif

#ifdef HAVE_TDM
// D# per CDNA5 ISA ch.8: group0 = {count/flags, lds_addr, global_addr, type=2},
// group1 = {data_size, tensor_dim0/1, tile_dim0/1/2, tensor_dim0_stride,...}.
__device__ __forceinline__ void tdm_load_2d_f16(u32 lds_off, const void* gptr,
                                                u32 tdim0, u32 tdim1,
                                                u32 tile0, u32 tile1,
                                                u32 stride_elems) {
  const u64 ga = (u64)gptr;
  v4u g0;
  g0.x = 1u;                                   // count=1, is_restore=0, no gather
  g0.y = lds_off;                              // lds_addr (bytes)
  g0.z = (u32)ga;                              // global_addr[31:0]
  g0.w = (u32)(ga >> 32) | 0x80000000u;        // global_addr[56:32] | type=2
  v8i g1;
  g1[0] = (int)0x00010000u;                    // data_size=1 (2 bytes/elem)
  g1[1] = (int)((tdim0 & 0xffffu) << 16);      // tensor_dim0[15:0]
  g1[2] = (int)((tdim0 >> 16) | ((tdim1 & 0xffffu) << 16));
  g1[3] = (int)((tdim1 >> 16) | (tile0 << 16)); // tile_dim0
  g1[4] = (int)(tile1 & 0xffffu);              // tile_dim1 (tile_dim2=0)
  g1[5] = (int)stride_elems;                   // tensor_dim0_stride[31:0]
  g1[6] = 0;                                   // stride[47:32], dim1_stride lo
  g1[7] = 0;
  v4i z4; z4[0] = 0; z4[1] = 0; z4[2] = 0; z4[3] = 0;
#if __clang_major__ >= 23
  v8i z8; for (int i = 0; i < 8; ++i) z8[i] = 0;
  __builtin_amdgcn_tensor_load_to_lds(g0, g1, z4, z4, z8, 0);
#else
  __builtin_amdgcn_tensor_load_to_lds(g0, g1, z4, z4, 0);
#endif
}
#endif

// ---- fp32 -> fp16 convert (grid-stride) ------------------------------------
__global__ void f32_to_f16_k(const float* __restrict__ in,
                             _Float16* __restrict__ out, size_t n) {
  size_t i = (size_t)blockIdx.x * 256 + threadIdx.x;
  const size_t stride = (size_t)gridDim.x * 256;
  for (; i < n; i += stride) out[i] = (_Float16)in[i];
}

// ---- RMSNorm (one 256-thread block per row of D_) -> f16 -------------------
__global__ __launch_bounds__(256)
void rmsnorm_f16(const float* __restrict__ x, const float* __restrict__ w,
                 _Float16* __restrict__ out) {
  __shared__ float red[8];
  const int row = blockIdx.x;
  const int tid = threadIdx.x;
  const float* xr = x + (size_t)row * D_;
  float ss = 0.f;
  for (int i = tid; i < D_; i += 256) { float v = xr[i]; ss += v * v; }
#pragma unroll
  for (int o = 16; o > 0; o >>= 1) ss += __shfl_xor(ss, o, 32);
  if ((tid & 31) == 0) red[tid >> 5] = ss;
  __syncthreads();
  if (tid == 0) {
    float t = 0.f;
    for (int i = 0; i < 8; ++i) t += red[i];
    red[0] = rsqrtf(t / (float)D_ + 1e-5f);
  }
  __syncthreads();
  const float rstd = red[0];
  _Float16* orow = out + (size_t)row * D_;
  for (int i = tid; i < D_; i += 256) orow[i] = (_Float16)(xr[i] * rstd * w[i]);
}

// ---- WMMA GEMM: C[M,N] = A[M,K](f16) * B[K,N](f16) + bias (+residual) ------
// block = 128 threads (4 waves), block tile 64x128, per-wave 32x64 (2x4 WMMA).
// A tile (64x32) staged via TDM async DMA (wave 0 issues; TENSORcnt waited),
// B tile (32x128) staged transposed by all threads.
template <bool OUT_F16>
__global__ __launch_bounds__(128)
void gemm_wmma(const _Float16* __restrict__ A, const _Float16* __restrict__ Bm,
               const float* __restrict__ bias, const float* __restrict__ residual,
               void* __restrict__ out, int M, int N, int K) {
  __shared__ _Float16 la[64 * 32];    // [m][k]
  __shared__ _Float16 lb[128 * 32];   // [n][k]  (transposed at stage time)
  const int tid  = threadIdx.x;
  const int lane = tid & 31;
  const int wv   = tid >> 5;
  const int l16  = lane & 15;
  const int lh   = (lane < 16) ? 0 : 1;   // lane half selects K sub-block
  const int wrow = (wv >> 1) * 32;
  const int wcol = (wv & 1) * 64;
  const int row0 = blockIdx.y * 64;
  const int col0 = blockIdx.x * 128;

  CF c[2][4];
#pragma unroll
  for (int mi = 0; mi < 2; ++mi)
#pragma unroll
    for (int ni = 0; ni < 4; ++ni)
#pragma unroll
      for (int t = 0; t < 8; ++t) c[mi][ni].f[t] = 0.f;

  const int arow = tid >> 1;          // 0..63
  const int acol = (tid & 1) * 16;    // 0 or 16
  const int bk   = tid & 31;          // 0..31
  const int bn   = (tid >> 5) * 16;   // 0,16,32,48

  for (int kt = 0; kt < K; kt += 32) {
#ifdef HAVE_TDM
    if (tid < 32) {   // one wave issues the DMA (TDM ignores EXEC)
      tdm_load_2d_f16((u32)(size_t)&la[0],
                      A + (size_t)row0 * K + kt,
                      (u32)K, (u32)M, 32u, 64u, (u32)K);
    }
#else
    {   // stage A tile 64x32 (row-major), 16 f16 per thread
      const u32* ga = (const u32*)(A + (size_t)(row0 + arow) * K + kt + acol);
      u32* lda = (u32*)&la[arow * 32 + acol];
#pragma unroll
      for (int j = 0; j < 8; ++j) lda[j] = ga[j];
    }
#endif
    {   // stage B tile 32x128 transposed into lb[n][k]
#pragma unroll
      for (int part = 0; part < 2; ++part) {
        const int nb = bn + part * 64;
        const _Float16* gb = Bm + (size_t)(kt + bk) * N + col0 + nb;
#pragma unroll
        for (int j = 0; j < 16; ++j) lb[(nb + j) * 32 + bk] = gb[j];
      }
    }
    if (kt + 32 < K) {   // gfx1250 global_prefetch_b8 on next tiles
      __builtin_prefetch(A + (size_t)(row0 + arow) * K + kt + 32 + acol, 0, 0);
      __builtin_prefetch(Bm + (size_t)(kt + 32 + bk) * N + col0 + bn, 0, 0);
    }
#ifdef HAVE_TDM
    if (tid < 32) __builtin_amdgcn_s_wait_tensorcnt(0);
#endif
    __syncthreads();

    AF af[2]; BFr bf[4];
#pragma unroll
    for (int mi = 0; mi < 2; ++mi) {   // A frag: row=l16, K contiguous per half
      const u32* pa = (const u32*)&la[(wrow + mi * 16 + l16) * 32];
#pragma unroll
      for (int j = 0; j < 8; ++j) af[mi].u[j] = pa[lh * 8 + j];
    }
#pragma unroll
    for (int ni = 0; ni < 4; ++ni) {   // B frag: col=l16, K contiguous per half
      const u32* pb = (const u32*)&lb[(wcol + ni * 16 + l16) * 32];
#pragma unroll
      for (int j = 0; j < 8; ++j) bf[ni].u[j] = pb[lh * 8 + j];
    }
#pragma unroll
    for (int mi = 0; mi < 2; ++mi)
#pragma unroll
      for (int ni = 0; ni < 4; ++ni)
        c[mi][ni].v = wmma16(af[mi].v, bf[ni].v, c[mi][ni].v);
    __syncthreads();
  }

  // epilogue: C layout lane=col, vgpr r = row r + 8*lh
#pragma unroll
  for (int mi = 0; mi < 2; ++mi) {
#pragma unroll
    for (int ni = 0; ni < 4; ++ni) {
      const int colg = col0 + wcol + ni * 16 + l16;
      const float bcol = bias[colg];
#pragma unroll
      for (int r = 0; r < 8; ++r) {
        const int rowg = row0 + wrow + mi * 16 + r + lh * 8;
        const size_t idx = (size_t)rowg * N + colg;
        float val = c[mi][ni].f[r] + bcol;
        if (residual) val += residual[idx];
        if (OUT_F16) ((_Float16*)out)[idx] = (_Float16)val;
        else         ((float*)out)[idx]    = val;
      }
    }
  }
}

// ---- RoPE on f16 activations [BT][NH*HD] in place --------------------------
__global__ void rope_f16(_Float16* __restrict__ t, const float* __restrict__ cs,
                         const float* __restrict__ sn, int NH) {
  const int p = blockIdx.x * 256 + threadIdx.x;
  const int total = BT_ * NH * (HD_ / 2);
  if (p >= total) return;
  const int i    = p % (HD_ / 2);
  const int hh   = (p / (HD_ / 2)) % NH;
  const int bt   = p / ((HD_ / 2) * NH);
  const int tpos = bt % T_;
  const size_t base = (size_t)bt * NH * HD_ + (size_t)hh * HD_ + 2 * i;
  const float x1 = (float)t[base], x2 = (float)t[base + 1];
  const float c = cs[tpos * (HD_ / 2) + i];
  const float s = sn[tpos * (HD_ / 2) + i];
  t[base]     = (_Float16)(x1 * c - x2 * s);
  t[base + 1] = (_Float16)(x1 * s + x2 * c);
}

// ---- Flash attention: 1 wave per (b, h, 16-row q tile) ---------------------
__global__ __launch_bounds__(32)
void attn_wmma(const _Float16* __restrict__ q, const _Float16* __restrict__ k,
               const _Float16* __restrict__ v, _Float16* __restrict__ ao) {
  __shared__ _Float16 p16[16 * 32];  // P tile staged for C->A transpose
  const int lane = threadIdx.x & 31;
  const int l16  = lane & 15;
  const int lh   = (lane < 16) ? 0 : 1;
  const int qb   = blockIdx.x * 16;
  const int hh   = blockIdx.y;
  const int b    = blockIdx.z;
  const int hk   = hh / (H_ / HKV_);
  const int bT   = b * T_;
  const float scale = 0.08838834764831845f;  // 1/sqrt(128)

  // Preload Q fragments for all 4 K-dim chunks (HD=128 = 4 x 32)
  AF qf[4];
  const u32* pq = (const u32*)(q + (size_t)(bT + qb + l16) * QD_ + hh * HD_);
#pragma unroll
  for (int c4 = 0; c4 < 4; ++c4) {
#pragma unroll
    for (int j = 0; j < 8; ++j) qf[c4].u[j] = pq[c4 * 16 + lh * 8 + j];
  }

  CF o[8];
#pragma unroll
  for (int dc = 0; dc < 8; ++dc)
#pragma unroll
    for (int r = 0; r < 8; ++r) o[dc].f[r] = 0.f;
  float mi[8], li[8];
#pragma unroll
  for (int r = 0; r < 8; ++r) { mi[r] = -1e30f; li[r] = 0.f; }

  for (int kb = 0; kb < qb + 16; kb += 32) {   // 32 keys per block
    CF s0, s1;
#pragma unroll
    for (int r = 0; r < 8; ++r) { s0.f[r] = 0.f; s1.f[r] = 0.f; }
    const u32* pr0 = (const u32*)(k + (size_t)(bT + kb + l16) * KVD_ + hk * HD_);
    const u32* pr1 = (const u32*)(k + (size_t)(bT + kb + 16 + l16) * KVD_ + hk * HD_);
#pragma unroll
    for (int c4 = 0; c4 < 4; ++c4) {
      BFr k0, k1;   // B frag: col = key index, contraction = head dim (contiguous)
#pragma unroll
      for (int j = 0; j < 8; ++j) {
        k0.u[j] = pr0[c4 * 16 + lh * 8 + j];
        k1.u[j] = pr1[c4 * 16 + lh * 8 + j];
      }
      s0.v = wmma16(qf[c4].v, k0.v, s0.v);
      s1.v = wmma16(qf[c4].v, k1.v, s1.v);
    }
    // scale + causal mask
#pragma unroll
    for (int r = 0; r < 8; ++r) {
      const int rowg = qb + r + lh * 8;
      float a = s0.f[r] * scale; if (kb + l16 > rowg)      a = -1e30f; s0.f[r] = a;
      float c = s1.f[r] * scale; if (kb + 16 + l16 > rowg) c = -1e30f; s1.f[r] = c;
    }
    // online softmax (row reductions across the 16-lane half holding the row)
    float alpha[8];
#pragma unroll
    for (int r = 0; r < 8; ++r) {
      float mloc = fmaxf(s0.f[r], s1.f[r]);
#pragma unroll
      for (int m = 1; m < 16; m <<= 1) mloc = fmaxf(mloc, __shfl_xor(mloc, m, 32));
      const float mn = fmaxf(mi[r], mloc);
      alpha[r] = expf(mi[r] - mn);
      const float p0 = expf(s0.f[r] - mn);
      const float p1 = expf(s1.f[r] - mn);
      s0.f[r] = p0; s1.f[r] = p1;
      float rs = p0 + p1;
#pragma unroll
      for (int m = 1; m < 16; m <<= 1) rs += __shfl_xor(rs, m, 32);
      li[r] = li[r] * alpha[r] + rs;
      mi[r] = mn;
    }
#pragma unroll
    for (int dc = 0; dc < 8; ++dc)
#pragma unroll
      for (int r = 0; r < 8; ++r) o[dc].f[r] *= alpha[r];

    // P (C layout) -> LDS -> A fragment
#pragma unroll
    for (int r = 0; r < 8; ++r) {
      p16[(r + lh * 8) * 32 + l16]      = (_Float16)s0.f[r];
      p16[(r + lh * 8) * 32 + 16 + l16] = (_Float16)s1.f[r];
    }
    __syncthreads();
    AF pf;
    const u32* pp = (const u32*)&p16[l16 * 32];
#pragma unroll
    for (int j = 0; j < 8; ++j) pf.u[j] = pp[lh * 8 + j];
    __syncthreads();

    // O += P(16x32) * V(32x128), 8 d-chunks of 16 cols
#pragma unroll
    for (int dc = 0; dc < 8; ++dc) {
      BFr vf;   // B frag: col = d, contraction = key (strided gather)
      const int d = dc * 16 + l16;
#pragma unroll
      for (int j = 0; j < 16; ++j)
        vf.h[j] = v[(size_t)(bT + kb + lh * 16 + j) * KVD_ + hk * HD_ + d];
      o[dc].v = wmma16(pf.v, vf.v, o[dc].v);
    }
  }

  // normalize + write f16
#pragma unroll
  for (int r = 0; r < 8; ++r) {
    const float inv = 1.f / li[r];
    const int rowg = qb + r + lh * 8;
    _Float16* orow = ao + (size_t)(bT + rowg) * QD_ + hh * HD_;
#pragma unroll
    for (int dc = 0; dc < 8; ++dc)
      orow[dc * 16 + l16] = (_Float16)(o[dc].f[r] * inv);
  }
}

// ---- SiLU(gate) * up -------------------------------------------------------
__global__ void silu_mul(const _Float16* __restrict__ gu,
                         _Float16* __restrict__ out) {
  const size_t idx = (size_t)blockIdx.x * 256 + threadIdx.x;
  const size_t total = (size_t)BT_ * I_;
  if (idx >= total) return;
  const size_t row = idx / I_, col = idx % I_;
  const float g = (float)gu[row * (size_t)(2 * I_) + col];
  const float u = (float)gu[row * (size_t)(2 * I_) + I_ + col];
  const float s = g / (1.f + expf(-g));
  out[row * (size_t)I_ + col] = (_Float16)(s * u);
}

// ---- workspace layout (bytes) ----------------------------------------------
static constexpr size_t OFF_WQ  = 0;
static constexpr size_t OFF_WK  = OFF_WQ  + (size_t)D_  * QD_      * 2;
static constexpr size_t OFF_WV  = OFF_WK  + (size_t)D_  * KVD_     * 2;
static constexpr size_t OFF_WO  = OFF_WV  + (size_t)D_  * KVD_     * 2;
static constexpr size_t OFF_W1  = OFF_WO  + (size_t)QD_ * D_       * 2;
static constexpr size_t OFF_W2  = OFF_W1  + (size_t)D_  * 2 * I_   * 2;
static constexpr size_t OFF_H   = OFF_W2  + (size_t)I_  * D_       * 2;
static constexpr size_t OFF_Q   = OFF_H   + (size_t)BT_ * D_       * 2;
static constexpr size_t OFF_K   = OFF_Q   + (size_t)BT_ * QD_      * 2;
static constexpr size_t OFF_V   = OFF_K   + (size_t)BT_ * KVD_     * 2;
static constexpr size_t OFF_AO  = OFF_V   + (size_t)BT_ * KVD_     * 2;
static constexpr size_t OFF_H1  = OFF_AO  + (size_t)BT_ * QD_      * 2;
static constexpr size_t OFF_G   = OFF_H1  + (size_t)BT_ * D_       * 4;
static constexpr size_t OFF_GU  = OFF_G   + (size_t)BT_ * D_       * 2;
static constexpr size_t OFF_MLP = OFF_GU  + (size_t)BT_ * 2 * I_   * 2;

extern "C" void kernel_launch(void* const* d_in, const int* in_sizes, int n_in,
                              void* d_out, int out_size, void* d_ws, size_t ws_size,
                              hipStream_t stream) {
  (void)in_sizes; (void)n_in; (void)out_size; (void)ws_size;
  const float* x        = (const float*)d_in[0];
  const float* rope_cos = (const float*)d_in[1];
  const float* rope_sin = (const float*)d_in[2];
  const float* ln1_w    = (const float*)d_in[3];
  const float* ln2_w    = (const float*)d_in[4];
  const float* wq_w     = (const float*)d_in[5];
  const float* wq_b     = (const float*)d_in[6];
  const float* wk_w     = (const float*)d_in[7];
  const float* wk_b     = (const float*)d_in[8];
  const float* wv_w     = (const float*)d_in[9];
  const float* wv_b     = (const float*)d_in[10];
  const float* wo_w     = (const float*)d_in[11];
  const float* wo_b     = (const float*)d_in[12];
  const float* w1_w     = (const float*)d_in[13];
  const float* w1_b     = (const float*)d_in[14];
  const float* w2_w     = (const float*)d_in[15];
  const float* w2_b     = (const float*)d_in[16];

  char* ws = (char*)d_ws;
  _Float16* wq16  = (_Float16*)(ws + OFF_WQ);
  _Float16* wk16  = (_Float16*)(ws + OFF_WK);
  _Float16* wv16  = (_Float16*)(ws + OFF_WV);
  _Float16* wo16  = (_Float16*)(ws + OFF_WO);
  _Float16* w116  = (_Float16*)(ws + OFF_W1);
  _Float16* w216  = (_Float16*)(ws + OFF_W2);
  _Float16* h16   = (_Float16*)(ws + OFF_H);
  _Float16* q16   = (_Float16*)(ws + OFF_Q);
  _Float16* k16   = (_Float16*)(ws + OFF_K);
  _Float16* v16   = (_Float16*)(ws + OFF_V);
  _Float16* ao16  = (_Float16*)(ws + OFF_AO);
  float*    h1f   = (float*)   (ws + OFF_H1);
  _Float16* g16   = (_Float16*)(ws + OFF_G);
  _Float16* gu16  = (_Float16*)(ws + OFF_GU);
  _Float16* mlp16 = (_Float16*)(ws + OFF_MLP);

  // 1) weights fp32 -> fp16
  f32_to_f16_k<<<2048, 256, 0, stream>>>(wq_w, wq16, (size_t)D_ * QD_);
  f32_to_f16_k<<<2048, 256, 0, stream>>>(wk_w, wk16, (size_t)D_ * KVD_);
  f32_to_f16_k<<<2048, 256, 0, stream>>>(wv_w, wv16, (size_t)D_ * KVD_);
  f32_to_f16_k<<<2048, 256, 0, stream>>>(wo_w, wo16, (size_t)QD_ * D_);
  f32_to_f16_k<<<2048, 256, 0, stream>>>(w1_w, w116, (size_t)D_ * 2 * I_);
  f32_to_f16_k<<<2048, 256, 0, stream>>>(w2_w, w216, (size_t)I_ * D_);

  // 2) RMSNorm1 -> h (f16)
  rmsnorm_f16<<<BT_, 256, 0, stream>>>(x, ln1_w, h16);

  // 3) QKV projections (WMMA, 64x128 tiles)
  gemm_wmma<true><<<dim3(QD_ / 128,  BT_ / 64), 128, 0, stream>>>(h16, wq16, wq_b, nullptr, q16, BT_, QD_,  D_);
  gemm_wmma<true><<<dim3(KVD_ / 128, BT_ / 64), 128, 0, stream>>>(h16, wk16, wk_b, nullptr, k16, BT_, KVD_, D_);
  gemm_wmma<true><<<dim3(KVD_ / 128, BT_ / 64), 128, 0, stream>>>(h16, wv16, wv_b, nullptr, v16, BT_, KVD_, D_);

  // 4) RoPE
  rope_f16<<<(BT_ * H_   * (HD_ / 2) + 255) / 256, 256, 0, stream>>>(q16, rope_cos, rope_sin, H_);
  rope_f16<<<(BT_ * HKV_ * (HD_ / 2) + 255) / 256, 256, 0, stream>>>(k16, rope_cos, rope_sin, HKV_);

  // 5) Flash attention (WMMA QK^T + PV)
  attn_wmma<<<dim3(T_ / 16, H_, B_), 32, 0, stream>>>(q16, k16, v16, ao16);

  // 6) WO projection + residual(x) -> h1 (f32)
  gemm_wmma<false><<<dim3(D_ / 128, BT_ / 64), 128, 0, stream>>>(ao16, wo16, wo_b, x, h1f, BT_, D_, QD_);

  // 7) RMSNorm2 -> g (f16)
  rmsnorm_f16<<<BT_, 256, 0, stream>>>(h1f, ln2_w, g16);

  // 8) W1 (gate_up) GEMM
  gemm_wmma<true><<<dim3((2 * I_) / 128, BT_ / 64), 128, 0, stream>>>(g16, w116, w1_b, nullptr, gu16, BT_, 2 * I_, D_);

  // 9) SiLU(gate) * up
  silu_mul<<<(int)(((size_t)BT_ * I_ + 255) / 256), 256, 0, stream>>>(gu16, mlp16);

  // 10) W2 GEMM + residual(h1) -> d_out (f32)
  gemm_wmma<false><<<dim3(D_ / 128, BT_ / 64), 128, 0, stream>>>(mlp16, w216, w2_b, h1f, (float*)d_out, BT_, D_, I_);
}
